// GATDecoder_64544768524829
// MI455X (gfx1250) — matrix operand order repaired and checked by reading, not compiled, not run on hardware
//
#include <hip/hip_runtime.h>

// ---------------- problem constants ----------------
static constexpr int HIDC   = 128;   // layer-0 input features
static constexpr int NHEADS = 3;
static constexpr int OPHC   = 43;    // channels per head (layers 0/1)
static constexpr int MIDC   = 129;   // NHEADS*OPHC
static constexpr int OUTCH  = 6;     // final output channels
static constexpr int CP     = 144;   // padded output cols (9 * 16)
static constexpr int KP     = 160;   // padded K (5 * 32)
static constexpr int THREADS = 256;

typedef __attribute__((ext_vector_type(16))) _Float16 v16h;
typedef __attribute__((ext_vector_type(8)))  _Float16 v8h;
typedef __attribute__((ext_vector_type(8)))  float    v8f;

__device__ __forceinline__ float leaky02(float x) { return x > 0.f ? x : 0.2f * x; }

// float atomic max via signed/unsigned bit ordering trick (works from -inf init)
__device__ __forceinline__ void atomicMaxF(float* a, float v) {
  if (v >= 0.f) atomicMax((int*)a, __float_as_int(v));
  else          atomicMin((unsigned int*)a, __float_as_uint(v));
}

__device__ __forceinline__ void edge_sd(const int* __restrict__ ei, int E, int e,
                                        int& s, int& d) {
  if (e < E) { s = ei[e]; d = ei[E + e]; }   // edge_index[0][e], edge_index[1][e]
  else       { s = d = e - E; }              // appended self-loop
}

// ---------------- conversion / init kernels ----------------
__global__ void k_cvt_x(const float* __restrict__ x, _Float16* __restrict__ xh,
                        int n_nodes, int rows, int kin) {
  int t = blockIdx.x * blockDim.x + threadIdx.x;
  if (t >= rows * KP) return;
  int n = t / KP, k = t - n * KP;
  xh[t] = (n < n_nodes && k < kin) ? (_Float16)x[n * kin + k] : (_Float16)0.f;
}

// WT[n][k] = W[k][n] (f16, zero padded) so B loads are contiguous in K
__global__ void k_cvt_wt(const float* __restrict__ W, _Float16* __restrict__ wt,
                         int kin, int cout) {
  int t = blockIdx.x * blockDim.x + threadIdx.x;
  if (t >= CP * KP) return;
  int n = t / KP, k = t - n * KP;
  wt[t] = (k < kin && n < cout) ? (_Float16)W[k * cout + n] : (_Float16)0.f;
}

__global__ void k_fill(float* __restrict__ p, float v, int n) {
  int t = blockIdx.x * blockDim.x + threadIdx.x;
  if (t < n) p[t] = v;
}

__global__ void k_fix_neginf(float* __restrict__ p, int n) {
  int t = blockIdx.x * blockDim.x + threadIdx.x;
  if (t >= n) return;
  float v = p[t];
  if (!(v >= -3.0e38f)) p[t] = 0.f;   // replace -inf (no in-edges) with 0
}

// ---------------- WMMA GEMM: H[M][CP] = Xh[M][KP] * WT^T ----------------
// One wave computes one 16x16 tile. Layouts per CDNA5 ISA 7.12.2:
//  A (16x32 f16): lane&15 = M row; lanes>=16 shift K by +8; VGPR4-7 hold K+16.
//  B (32x16 f16): lane&15 = N col; lanes>=16 shift K by +8; elements 8-15 hold K+16.
//  C/D (16x16 f32): VGPR r = row r (lanes 0-15) / row 8+r (lanes 16-31), N = lane&15.
__global__ void __launch_bounds__(THREADS)
k_gemm_wmma(const _Float16* __restrict__ Xh, const _Float16* __restrict__ WT,
            float* __restrict__ H, int mtiles, int kchunks) {
  int wave = (blockIdx.x * blockDim.x + threadIdx.x) >> 5;
  int lane = threadIdx.x & 31;
  const int ntiles = CP / 16;
  int tile_m = wave / ntiles;
  int tile_n = wave - tile_m * ntiles;
  if (tile_m >= mtiles) return;

  int l15 = lane & 15;
  int hi  = (lane >> 4) & 1;
  const _Float16* aRow = Xh + (size_t)(tile_m * 16 + l15) * KP;
  const _Float16* bRow = WT + (size_t)(tile_n * 16 + l15) * KP;

  v8f acc = {};
  for (int kc = 0; kc < kchunks; ++kc) {
    int koff = kc * 32;
    v8h alo = *(const v8h*)(aRow + koff + hi * 8);
    v8h ahi = *(const v8h*)(aRow + koff + hi * 8 + 16);
    v16h a;
#pragma unroll
    for (int i = 0; i < 8; ++i) { a[i] = alo[i]; a[i + 8] = ahi[i]; }
    v8h blo = *(const v8h*)(bRow + koff + hi * 8);
    v8h bhi = *(const v8h*)(bRow + koff + hi * 8 + 16);
    v16h b;
#pragma unroll
    for (int i = 0; i < 8; ++i) { b[i] = blo[i]; b[i + 8] = bhi[i]; }
    acc = __builtin_amdgcn_wmma_f32_16x16x32_f16(false, a, false, b,
                                                 (short)0, acc, false, false);
  }

  int ncol  = tile_n * 16 + l15;
  int mbase = tile_m * 16 + hi * 8;
#pragma unroll
  for (int r = 0; r < 8; ++r) H[(size_t)(mbase + r) * CP + ncol] = acc[r];
}

// ---------------- attention kernels ----------------
__global__ void k_alar(const float* __restrict__ H, const float* __restrict__ as,
                       const float* __restrict__ ad, float* __restrict__ al,
                       float* __restrict__ ar, int n_nodes) {
  int t = blockIdx.x * blockDim.x + threadIdx.x;
  if (t >= n_nodes * NHEADS) return;
  int n = t / NHEADS, hd = t - n * NHEADS;
  const float* hp = H + (size_t)n * CP + hd * OPHC;
  const float* sp = as + hd * OPHC;
  const float* dp = ad + hd * OPHC;
  float s = 0.f, d = 0.f;
#pragma unroll 1
  for (int c = 0; c < OPHC; ++c) { float v = hp[c]; s += v * sp[c]; d += v * dp[c]; }
  al[t] = s; ar[t] = d;
}

__global__ void k_edge_max(const int* __restrict__ ei, int E, int EA,
                           const float* __restrict__ al, const float* __restrict__ ar,
                           float* __restrict__ emax, int heads) {
  int t = blockIdx.x * blockDim.x + threadIdx.x;
  if (t >= EA * heads) return;
  int e = t / heads, hd = t - e * heads;
  int s, d; edge_sd(ei, E, e, s, d);
  float v = leaky02(al[s * heads + hd] + ar[d * heads + hd]);
  atomicMaxF(&emax[d * heads + hd], v);
}

__global__ void k_edge_den(const int* __restrict__ ei, int E, int EA,
                           const float* __restrict__ al, const float* __restrict__ ar,
                           const float* __restrict__ emax, float* __restrict__ den,
                           int heads) {
  int t = blockIdx.x * blockDim.x + threadIdx.x;
  if (t >= EA * heads) return;
  int e = t / heads, hd = t - e * heads;
  int s, d; edge_sd(ei, E, e, s, d);
  float v  = leaky02(al[s * heads + hd] + ar[d * heads + hd]);
  float ex = __expf(v - emax[d * heads + hd]);
  atomicAdd(&den[d * heads + hd], ex);
}

__global__ void k_edge_msg(const int* __restrict__ ei, int E, int EA,
                           const float* __restrict__ al, const float* __restrict__ ar,
                           const float* __restrict__ emax, const float* __restrict__ den,
                           const float* __restrict__ H, float* __restrict__ agg,
                           int heads, int ch, int hstride) {
  int t = blockIdx.x * blockDim.x + threadIdx.x;
  if (t >= EA * heads) return;
  int e = t / heads, hd = t - e * heads;
  int s, d; edge_sd(ei, E, e, s, d);
  float v     = leaky02(al[s * heads + hd] + ar[d * heads + hd]);
  float ex    = __expf(v - emax[d * heads + hd]);
  float alpha = ex / (den[d * heads + hd] + 1e-16f);
  const float* hp = H + (size_t)s * hstride + hd * ch;
  float*       op = agg + (size_t)d * hstride + hd * ch;
#pragma unroll 1
  for (int c = 0; c < ch; ++c) atomicAdd(&op[c], hp[c] * alpha);
}

// out = leaky(agg + b), in place (f32), plus f16 re-pack for next GEMM
__global__ void k_finalize(float* __restrict__ agg, const float* __restrict__ b,
                           _Float16* __restrict__ xh, int n_nodes) {
  int t = blockIdx.x * blockDim.x + threadIdx.x;
  if (t >= n_nodes * KP) return;
  int n = t / KP, c = t - n * KP;
  if (c < MIDC) {
    float v = leaky02(agg[(size_t)n * CP + c] + b[c]);
    agg[(size_t)n * CP + c] = v;
    xh[t] = (_Float16)v;
  } else {
    xh[t] = (_Float16)0.f;
  }
}

// ---------------- tiny layer-2 kernels (129 -> 6, 1 head) ----------------
__global__ void k_gemm2(const float* __restrict__ Y, const float* __restrict__ W2,
                        float* __restrict__ H2, int n_nodes) {
  int t = blockIdx.x * blockDim.x + threadIdx.x;
  if (t >= n_nodes * OUTCH) return;
  int n = t / OUTCH, c = t - n * OUTCH;
  const float* yp = Y + (size_t)n * CP;
  float s = 0.f;
#pragma unroll 1
  for (int k = 0; k < MIDC; ++k) s += yp[k] * W2[k * OUTCH + c];
  H2[t] = s;
}

__global__ void k_alar2(const float* __restrict__ H2, const float* __restrict__ as,
                        const float* __restrict__ ad, float* __restrict__ al,
                        float* __restrict__ ar, int n_nodes) {
  int n = blockIdx.x * blockDim.x + threadIdx.x;
  if (n >= n_nodes) return;
  float s = 0.f, d = 0.f;
#pragma unroll
  for (int c = 0; c < OUTCH; ++c) {
    float v = H2[(size_t)n * OUTCH + c];
    s += v * as[c]; d += v * ad[c];
  }
  al[n] = s; ar[n] = d;
}

__global__ void k_out(const float* __restrict__ agg2, const float* __restrict__ b2,
                      float* __restrict__ out, int n_nodes) {
  int t = blockIdx.x * blockDim.x + threadIdx.x;
  if (t >= n_nodes * OUTCH) return;
  out[t] = agg2[t] + b2[t % OUTCH];
}

// ---------------- host launcher ----------------
static inline unsigned gridFor(long long n) {
  return (unsigned)((n + THREADS - 1) / THREADS);
}

extern "C" void kernel_launch(void* const* d_in, const int* in_sizes, int n_in,
                              void* d_out, int out_size, void* d_ws, size_t ws_size,
                              hipStream_t stream) {
  const float* x   = (const float*)d_in[0];
  const int*   ei  = (const int*)d_in[1];
  const float* W0  = (const float*)d_in[2];
  const float* a0s = (const float*)d_in[3];
  const float* a0d = (const float*)d_in[4];
  const float* b0  = (const float*)d_in[5];
  const float* W1  = (const float*)d_in[6];
  const float* a1s = (const float*)d_in[7];
  const float* a1d = (const float*)d_in[8];
  const float* b1  = (const float*)d_in[9];
  const float* W2  = (const float*)d_in[10];
  const float* a2s = (const float*)d_in[11];
  const float* a2d = (const float*)d_in[12];
  const float* b2  = (const float*)d_in[13];
  float* out = (float*)d_out;
  (void)n_in; (void)out_size; (void)ws_size;

  const int n_nodes = in_sizes[0] / HIDC;       // 50000
  const int E       = in_sizes[1] / 2;          // 800000
  const int EA      = E + n_nodes;              // with self-loops
  const int mt      = (n_nodes + 15) / 16;      // M tiles (50000/16 = 3125)
  const int MT16    = mt * 16;

  // scratch carving (everything stays L2-resident: ~80 MB total)
  char* ws = (char*)d_ws;
  size_t off = 0;
  auto carve = [&](size_t bytes) {
    void* p = ws + off;
    off += (bytes + 255) & ~(size_t)255;
    return p;
  };
  _Float16* XH  = (_Float16*)carve((size_t)MT16 * KP * sizeof(_Float16));
  _Float16* WT  = (_Float16*)carve((size_t)CP * KP * sizeof(_Float16));
  float* H    = (float*)carve((size_t)MT16 * CP * sizeof(float));
  float* AGG  = (float*)carve((size_t)MT16 * CP * sizeof(float));
  float* AL   = (float*)carve((size_t)n_nodes * NHEADS * sizeof(float));
  float* AR   = (float*)carve((size_t)n_nodes * NHEADS * sizeof(float));
  float* EMX  = (float*)carve((size_t)n_nodes * NHEADS * sizeof(float));
  float* DEN  = (float*)carve((size_t)n_nodes * NHEADS * sizeof(float));
  float* AGG2 = (float*)carve((size_t)n_nodes * OUTCH * sizeof(float));

  const float NEG_INF = -__builtin_inff();
  const long long gemmThreads = (long long)mt * (CP / 16) * 32;

  // ================= Layer 0 =================
  k_cvt_x <<<gridFor((long long)MT16 * KP), THREADS, 0, stream>>>(x, XH, n_nodes, MT16, HIDC);
  k_cvt_wt<<<gridFor(CP * KP), THREADS, 0, stream>>>(W0, WT, HIDC, MIDC);
  k_gemm_wmma<<<gridFor(gemmThreads), THREADS, 0, stream>>>(XH, WT, H, mt, HIDC / 32);
  k_alar<<<gridFor((long long)n_nodes * NHEADS), THREADS, 0, stream>>>(H, a0s, a0d, AL, AR, n_nodes);
  k_fill<<<gridFor((long long)n_nodes * NHEADS), THREADS, 0, stream>>>(EMX, NEG_INF, n_nodes * NHEADS);
  k_fill<<<gridFor((long long)n_nodes * NHEADS), THREADS, 0, stream>>>(DEN, 0.f, n_nodes * NHEADS);
  k_fill<<<gridFor((long long)MT16 * CP), THREADS, 0, stream>>>(AGG, 0.f, MT16 * CP);
  k_edge_max<<<gridFor((long long)EA * NHEADS), THREADS, 0, stream>>>(ei, E, EA, AL, AR, EMX, NHEADS);
  k_fix_neginf<<<gridFor((long long)n_nodes * NHEADS), THREADS, 0, stream>>>(EMX, n_nodes * NHEADS);
  k_edge_den<<<gridFor((long long)EA * NHEADS), THREADS, 0, stream>>>(ei, E, EA, AL, AR, EMX, DEN, NHEADS);
  k_edge_msg<<<gridFor((long long)EA * NHEADS), THREADS, 0, stream>>>(ei, E, EA, AL, AR, EMX, DEN, H, AGG, NHEADS, OPHC, CP);
  k_finalize<<<gridFor((long long)n_nodes * KP), THREADS, 0, stream>>>(AGG, b0, XH, n_nodes);

  // ================= Layer 1 =================
  k_cvt_wt<<<gridFor(CP * KP), THREADS, 0, stream>>>(W1, WT, MIDC, MIDC);
  k_gemm_wmma<<<gridFor(gemmThreads), THREADS, 0, stream>>>(XH, WT, H, mt, (MIDC + 31) / 32);
  k_alar<<<gridFor((long long)n_nodes * NHEADS), THREADS, 0, stream>>>(H, a1s, a1d, AL, AR, n_nodes);
  k_fill<<<gridFor((long long)n_nodes * NHEADS), THREADS, 0, stream>>>(EMX, NEG_INF, n_nodes * NHEADS);
  k_fill<<<gridFor((long long)n_nodes * NHEADS), THREADS, 0, stream>>>(DEN, 0.f, n_nodes * NHEADS);
  k_fill<<<gridFor((long long)MT16 * CP), THREADS, 0, stream>>>(AGG, 0.f, MT16 * CP);
  k_edge_max<<<gridFor((long long)EA * NHEADS), THREADS, 0, stream>>>(ei, E, EA, AL, AR, EMX, NHEADS);
  k_fix_neginf<<<gridFor((long long)n_nodes * NHEADS), THREADS, 0, stream>>>(EMX, n_nodes * NHEADS);
  k_edge_den<<<gridFor((long long)EA * NHEADS), THREADS, 0, stream>>>(ei, E, EA, AL, AR, EMX, DEN, NHEADS);
  k_edge_msg<<<gridFor((long long)EA * NHEADS), THREADS, 0, stream>>>(ei, E, EA, AL, AR, EMX, DEN, H, AGG, NHEADS, OPHC, CP);
  k_finalize<<<gridFor((long long)n_nodes * KP), THREADS, 0, stream>>>(AGG, b1, XH, n_nodes);
  // AGG now holds y2 = leaky(layer1 out + b1) in f32, stride CP

  // ================= Layer 2 (1 head, 6 ch, mean==identity) =================
  k_gemm2<<<gridFor((long long)n_nodes * OUTCH), THREADS, 0, stream>>>(AGG, W2, H, n_nodes); // reuse H as h2 [N][6]
  k_alar2<<<gridFor((long long)n_nodes), THREADS, 0, stream>>>(H, a2s, a2d, AL, AR, n_nodes);
  k_fill<<<gridFor((long long)n_nodes), THREADS, 0, stream>>>(EMX, NEG_INF, n_nodes);
  k_fill<<<gridFor((long long)n_nodes), THREADS, 0, stream>>>(DEN, 0.f, n_nodes);
  k_fill<<<gridFor((long long)n_nodes * OUTCH), THREADS, 0, stream>>>(AGG2, 0.f, n_nodes * OUTCH);
  k_edge_max<<<gridFor((long long)EA), THREADS, 0, stream>>>(ei, E, EA, AL, AR, EMX, 1);
  k_fix_neginf<<<gridFor((long long)n_nodes), THREADS, 0, stream>>>(EMX, n_nodes);
  k_edge_den<<<gridFor((long long)EA), THREADS, 0, stream>>>(ei, E, EA, AL, AR, EMX, DEN, 1);
  k_edge_msg<<<gridFor((long long)EA), THREADS, 0, stream>>>(ei, E, EA, AL, AR, EMX, DEN, H, AGG2, 1, OUTCH, OUTCH);
  k_out<<<gridFor((long long)n_nodes * OUTCH), THREADS, 0, stream>>>(AGG2, b2, out, n_nodes);
}
